// CausalSelfAttentionImpl_48541720380012
// MI455X (gfx1250) — matrix operand order repaired and checked by reading, not compiled, not run on hardware
//
#include <hip/hip_runtime.h>
#include <hip/hip_bf16.h>

// CDNA5 / gfx1250 causal self-attention forward.
//   1) cast x, w_qkv, w_proj fp32 -> bf16 (workspace)
//   2) QKV GEMM (wmma bf16, 32x64 tiles/wave): q,k,v bf16 as [B,H,T,D]
//   3) flash attention: wmma bf16 QK^T from global, V tile staged to LDS by the
//      Tensor Data Mover (tensor_load_to_lds, TENSORcnt), wmma bf16 PV from LDS
//   4) output projection (wmma bf16, 32x64 tiles/wave, fp32 out)

typedef __bf16 bf16;
typedef __attribute__((ext_vector_type(16))) __bf16 v16bf;
typedef __attribute__((ext_vector_type(8)))  float  v8f;
typedef __attribute__((ext_vector_type(4)))  unsigned int u32x4;
typedef __attribute__((ext_vector_type(8)))  int i32x8;
typedef __attribute__((ext_vector_type(4)))  int i32x4;

#define B_   2
#define T_   2048
#define C_   1024
#define H_   16
#define D_   64
#define BT_  (B_ * T_)

static __device__ __forceinline__ v8f wmma_bf16(v16bf a, v16bf b, v8f c) {
  return __builtin_amdgcn_wmma_f32_16x16x32_bf16(false, a, false, b, (short)0, c,
                                                 false, false);
}

union V16U { v16bf v; uint4 q[2]; };

// A-matrix 16x32 bf16 tile (contiguous k order, same as the B loads): lane L holds row = row_base + (L&15);
// element e -> k = k_base + e + 16*(L>=16).
static __device__ __forceinline__ v16bf load_A(const bf16* base, int ld,
                                               int row_base, int k_base, int lane) {
  int r  = row_base + (lane & 15);
  int hi = (lane >> 4) & 1;
  const bf16* p = base + (size_t)r * ld + k_base + 16 * hi;
  V16U u;
  u.q[0] = *(const uint4*)(p);        // e = 0..7
  u.q[1] = *(const uint4*)(p + 8);    // e = 8..15
  return u.v;
}

// B-matrix 32x16 bf16 tile where column n of B is row n of row-major M (X @ M^T):
// lane L holds column n = col + (L&15); element e -> k = k_base + e + 16*(L>=16).
static __device__ __forceinline__ v16bf load_B_rowmajorT(const bf16* base, int ld,
                                                         int k_base, int col, int lane) {
  int n  = col + (lane & 15);
  int hi = (lane >> 4) & 1;
  const bf16* p = base + (size_t)n * ld + k_base + 16 * hi;
  V16U u;
  u.q[0] = *(const uint4*)(p);
  u.q[1] = *(const uint4*)(p + 8);
  return u.v;
}

// B-matrix 32x16 bf16 tile with strided contraction axis (P @ V, V row-major
// [key][d], contraction over key): element e at base[(k_base+e+16*hi)*kstride + col+(L&15)].
static __device__ __forceinline__ v16bf load_B_strided(const bf16* base, int kstride,
                                                       int k_base, int col, int lane) {
  int n  = col + (lane & 15);
  int hi = (lane >> 4) & 1;
  const bf16* p = base + (size_t)(k_base + 16 * hi) * kstride + n;
  v16bf b;
#pragma unroll
  for (int e = 0; e < 16; ++e) b[e] = p[(size_t)e * kstride];
  return b;
}

// Generic pointer to a __shared__ object -> raw LDS byte offset (AS3 addresses
// are the low 32 bits of the generic LDS aperture address).
static __device__ __forceinline__ unsigned lds_off(void* p) {
  return (unsigned)(uintptr_t)(__attribute__((address_space(3))) void*)p;
}

// Issue one TDM 2D tile load: tensor == tile (no OOB), bf16 elements.
// D# layout per CDNA5 ISA.
static __device__ __forceinline__ void tdm_load_2d_bf16(unsigned lds_addr,
                                                        const void* gaddr,
                                                        unsigned dim0, unsigned dim1,
                                                        unsigned stride0) {
  unsigned long long ga = (unsigned long long)(uintptr_t)gaddr;
  u32x4 g0;
  g0[0] = 1u;                                            // count=1, user D#
  g0[1] = lds_addr;                                      // lds_addr [63:32]
  g0[2] = (unsigned)(ga & 0xffffffffu);                  // global_addr lo
  g0[3] = (unsigned)((ga >> 32) & 0x01ffffffu) | (2u << 30);  // addr hi | type=2
  i32x8 g1;
  g1[0] = (int)(1u << 16);                               // wg_mask=0, data_size=1 (2B)
  g1[1] = (int)((dim0 & 0xffffu) << 16);                 // tensor_dim0 lo16
  g1[2] = (int)(((dim0 >> 16) & 0xffffu) | ((dim1 & 0xffffu) << 16));
  g1[3] = (int)(((dim1 >> 16) & 0xffffu) | ((dim0 & 0xffffu) << 16));  // tile_dim0 = dim0
  g1[4] = (int)(dim1 & 0xffffu);                         // tile_dim1 = dim1, tile_dim2=0
  g1[5] = (int)stride0;                                  // tensor_dim0_stride lo32
  g1[6] = 0;
  g1[7] = 0;
  i32x4 z4 = {0, 0, 0, 0};
#if defined(__clang_major__) && (__clang_major__ >= 23)
  i32x8 z8 = {0, 0, 0, 0, 0, 0, 0, 0};
  __builtin_amdgcn_tensor_load_to_lds(g0, g1, z4, z4, z8, 0);
#else
  __builtin_amdgcn_tensor_load_to_lds(g0, g1, z4, z4, 0);
#endif
}

__global__ void cast_f32_to_bf16_kernel(const float* __restrict__ in,
                                        bf16* __restrict__ out, int n) {
  int i = blockIdx.x * blockDim.x + threadIdx.x;
  if (i < n) out[i] = (bf16)in[i];
}

// y = x @ w_qkv^T -> q/k/v bf16 [B,H,T,D]. One wave: 32(M) x 64(N) tile.
__global__ __launch_bounds__(32)
void qkv_gemm_kernel(const bf16* __restrict__ xb, const bf16* __restrict__ wb,
                     bf16* __restrict__ q, bf16* __restrict__ k, bf16* __restrict__ v) {
  const int lane   = threadIdx.x;
  const int mtiles = BT_ / 32;
  const int mt = blockIdx.x % mtiles;
  const int nt = blockIdx.x / mtiles;   // 48 tiles of 64 cols over 3C

  v8f acc[2][4] = {};
  for (int kk = 0; kk < C_; kk += 32) {
    v16bf a0 = load_A(xb, C_, mt * 32, kk, lane);
    v16bf a1 = load_A(xb, C_, mt * 32 + 16, kk, lane);
#pragma unroll
    for (int j = 0; j < 4; ++j) {
      v16bf b = load_B_rowmajorT(wb, C_, kk, nt * 64 + j * 16, lane);
      acc[0][j] = wmma_bf16(a0, b, acc[0][j]);
      acc[1][j] = wmma_bf16(a1, b, acc[1][j]);
    }
  }
  const int hi = (lane >> 4) & 1;
#pragma unroll
  for (int r = 0; r < 2; ++r) {
#pragma unroll
    for (int j = 0; j < 4; ++j) {
#pragma unroll
      for (int i = 0; i < 8; ++i) {
        int m = mt * 32 + r * 16 + i + 8 * hi;
        int n = nt * 64 + j * 16 + (lane & 15);
        int bb = m / T_, t = m % T_;
        int sel = n / C_, c = n % C_;
        int h = c / D_, d = c % D_;
        bf16* dst = (sel == 0) ? q : ((sel == 1) ? k : v);
        dst[(((size_t)bb * H_ + h) * T_ + t) * D_ + d] = (bf16)acc[r][j][i];
      }
    }
  }
}

// Flash attention: one wave per (b, h, 16-query tile). Key blocks of 32.
// V tile staged into LDS by the Tensor Data Mover, overlapped with QK^T.
__global__ __launch_bounds__(32)
void attn_kernel(const bf16* __restrict__ q, const bf16* __restrict__ k,
                 const bf16* __restrict__ v, bf16* __restrict__ ao) {
  __shared__ __align__(16) bf16 ldsV[32 * 64];   // 4 KB V tile
  __shared__ __align__(16) bf16 ldsP[16 * 32];   // 1 KB P tile

  const int lane   = threadIdx.x;
  const int qtiles = T_ / 16;
  const int qt = blockIdx.x % qtiles;
  const int h  = (blockIdx.x / qtiles) % H_;
  const int bb = blockIdx.x / (qtiles * H_);

  const bf16* qp = q + ((size_t)bb * H_ + h) * T_ * D_;
  const bf16* kp = k + ((size_t)bb * H_ + h) * T_ * D_;
  const bf16* vp = v + ((size_t)bb * H_ + h) * T_ * D_;

  const int q0 = qt * 16;
  const int hi = (lane >> 4) & 1;
  const float scale = 0.125f;  // 1/sqrt(64)
  const unsigned ldsV_addr = lds_off(ldsV);

  v16bf aq0 = load_A(qp, D_, q0, 0, lane);
  v16bf aq1 = load_A(qp, D_, q0, 32, lane);

  v8f o[4] = {};
  float mrow[8], lrow[8];
#pragma unroll
  for (int i = 0; i < 8; ++i) { mrow[i] = -1e30f; lrow[i] = 0.0f; }

  for (int kb = 0; kb <= q0 + 15; kb += 32) {   // uniform bound, EXEC stays full
    // WAR guard: previous iteration's LDS reads must drain before TDM rewrites
    // the V tile (the compiler cannot see the TDM<->ds hazard).
    __asm__ __volatile__("s_wait_dscnt 0x0" ::: "memory");
    // Async DMA: V rows [kb, kb+32) x 64 -> LDS, overlapped with S = Q K^T.
    tdm_load_2d_bf16(ldsV_addr, vp + (size_t)kb * D_, D_, 32, D_);

    // S = Q K^T for keys [kb, kb+32): two 16x16 score tiles from global K.
    v8f s0 = {}, s1 = {};
    {
      v16bf b;
      b = load_B_rowmajorT(kp, D_, 0,  kb,      lane); s0 = wmma_bf16(aq0, b, s0);
      b = load_B_rowmajorT(kp, D_, 32, kb,      lane); s0 = wmma_bf16(aq1, b, s0);
      b = load_B_rowmajorT(kp, D_, 0,  kb + 16, lane); s1 = wmma_bf16(aq0, b, s1);
      b = load_B_rowmajorT(kp, D_, 32, kb + 16, lane); s1 = wmma_bf16(aq1, b, s1);
    }
    // Scale + causal mask (selects only; no divergence).
    const int key0 = kb + (lane & 15);
#pragma unroll
    for (int i = 0; i < 8; ++i) {
      int m = q0 + i + 8 * hi;
      float a0 = s0[i] * scale; if (key0 > m)      a0 = -1e30f;
      float a1 = s1[i] * scale; if (key0 + 16 > m) a1 = -1e30f;
      s0[i] = a0; s1[i] = a1;
    }
    // Online softmax; rows live across 16-lane half-groups (xor 1/2/4/8).
    float alpha[8];
#pragma unroll
    for (int i = 0; i < 8; ++i) {
      float rm = fmaxf(s0[i], s1[i]);
      rm = fmaxf(rm, __shfl_xor(rm, 1, 32));
      rm = fmaxf(rm, __shfl_xor(rm, 2, 32));
      rm = fmaxf(rm, __shfl_xor(rm, 4, 32));
      rm = fmaxf(rm, __shfl_xor(rm, 8, 32));
      float mnew = fmaxf(mrow[i], rm);
      alpha[i] = __expf(mrow[i] - mnew);
      mrow[i] = mnew;
      float p0 = __expf(s0[i] - mnew);
      float p1 = __expf(s1[i] - mnew);
      s0[i] = p0; s1[i] = p1;
      float rs = p0 + p1;
      rs += __shfl_xor(rs, 1, 32);
      rs += __shfl_xor(rs, 2, 32);
      rs += __shfl_xor(rs, 4, 32);
      rs += __shfl_xor(rs, 8, 32);
      lrow[i] = lrow[i] * alpha[i] + rs;
    }
#pragma unroll
    for (int j = 0; j < 4; ++j)
#pragma unroll
      for (int i = 0; i < 8; ++i) o[j][i] *= alpha[i];

    // D-layout P -> row-major 16x32 bf16 in LDS, reload in A layout.
#pragma unroll
    for (int i = 0; i < 8; ++i) {
      ldsP[(i + 8 * hi) * 32 + (lane & 15)]      = (bf16)s0[i];
      ldsP[(i + 8 * hi) * 32 + 16 + (lane & 15)] = (bf16)s1[i];
    }
    __syncthreads();
    v16bf ap = load_A(ldsP, 32, 0, 0, lane);

    // Wait for the TDM V tile, then O += P @ V from LDS.
    __builtin_amdgcn_s_wait_tensorcnt(0);
    __asm__ __volatile__("" ::: "memory");
#pragma unroll
    for (int j = 0; j < 4; ++j) {
      v16bf b = load_B_strided(ldsV, D_, 0, j * 16, lane);
      o[j] = wmma_bf16(ap, b, o[j]);
    }
    __syncthreads();
  }

  // Normalize and store attention output as [B,T,C] bf16.
#pragma unroll
  for (int j = 0; j < 4; ++j)
#pragma unroll
    for (int i = 0; i < 8; ++i) {
      int m = q0 + i + 8 * hi;
      int d = j * 16 + (lane & 15);
      float val = o[j][i] / lrow[i];
      ao[((size_t)bb * T_ + m) * C_ + h * D_ + d] = (bf16)val;
    }
}

// out = attn_out @ w_proj^T, fp32 output. One wave: 32(M) x 64(N) tile.
__global__ __launch_bounds__(32)
void proj_gemm_kernel(const bf16* __restrict__ ab, const bf16* __restrict__ wb,
                      float* __restrict__ out) {
  const int lane   = threadIdx.x;
  const int mtiles = BT_ / 32;
  const int mt = blockIdx.x % mtiles;
  const int nt = blockIdx.x / mtiles;   // 16 tiles of 64 over C

  v8f acc[2][4] = {};
  for (int kk = 0; kk < C_; kk += 32) {
    v16bf a0 = load_A(ab, C_, mt * 32, kk, lane);
    v16bf a1 = load_A(ab, C_, mt * 32 + 16, kk, lane);
#pragma unroll
    for (int j = 0; j < 4; ++j) {
      v16bf b = load_B_rowmajorT(wb, C_, kk, nt * 64 + j * 16, lane);
      acc[0][j] = wmma_bf16(a0, b, acc[0][j]);
      acc[1][j] = wmma_bf16(a1, b, acc[1][j]);
    }
  }
  const int hi = (lane >> 4) & 1;
#pragma unroll
  for (int r = 0; r < 2; ++r)
#pragma unroll
    for (int j = 0; j < 4; ++j)
#pragma unroll
      for (int i = 0; i < 8; ++i) {
        int m = mt * 32 + r * 16 + i + 8 * hi;
        int n = nt * 64 + j * 16 + (lane & 15);
        out[(size_t)m * C_ + n] = acc[r][j][i];
      }
}

extern "C" void kernel_launch(void* const* d_in, const int* in_sizes, int n_in,
                              void* d_out, int out_size, void* d_ws, size_t ws_size,
                              hipStream_t stream) {
  const float* x      = (const float*)d_in[0];
  const float* w_qkv  = (const float*)d_in[1];
  const float* w_proj = (const float*)d_in[2];
  float* out = (float*)d_out;

  char* ws = (char*)d_ws;
  size_t off = 0;
  bf16* xb     = (bf16*)(ws + off); off += (size_t)BT_ * C_ * sizeof(bf16);      // 8 MB
  bf16* wqkvb  = (bf16*)(ws + off); off += (size_t)3 * C_ * C_ * sizeof(bf16);   // 6 MB
  bf16* wprojb = (bf16*)(ws + off); off += (size_t)C_ * C_ * sizeof(bf16);       // 2 MB
  bf16* qb     = (bf16*)(ws + off); off += (size_t)BT_ * C_ * sizeof(bf16);      // 8 MB
  bf16* kb     = (bf16*)(ws + off); off += (size_t)BT_ * C_ * sizeof(bf16);      // 8 MB
  bf16* vb     = (bf16*)(ws + off); off += (size_t)BT_ * C_ * sizeof(bf16);      // 8 MB
  bf16* ao     = (bf16*)(ws + off); off += (size_t)BT_ * C_ * sizeof(bf16);      // 8 MB
  (void)ws_size; (void)in_sizes; (void)n_in; (void)out_size;

  const int nx = BT_ * C_;
  const int nw = 3 * C_ * C_;
  const int np = C_ * C_;
  cast_f32_to_bf16_kernel<<<(nx + 255) / 256, 256, 0, stream>>>(x, xb, nx);
  cast_f32_to_bf16_kernel<<<(nw + 255) / 256, 256, 0, stream>>>(w_qkv, wqkvb, nw);
  cast_f32_to_bf16_kernel<<<(np + 255) / 256, 256, 0, stream>>>(w_proj, wprojb, np);

  qkv_gemm_kernel<<<(BT_ / 32) * (3 * C_ / 64), 32, 0, stream>>>(xb, wqkvb, qb, kb, vb);
  attn_kernel<<<B_ * H_ * (T_ / 16), 32, 0, stream>>>(qb, kb, vb, ao);
  proj_gemm_kernel<<<(BT_ / 32) * (C_ / 64), 32, 0, stream>>>(ao, wprojb, out);
}
